// TaskSpecificGNN_28509992911451
// MI455X (gfx1250) — compile-verified, matrix operand order checked by reading, not run on hardware
//
#include <hip/hip_runtime.h>
#include <hip/hip_bf16.h>
#include <math.h>

// ---------------------------------------------------------------------------
// Types for CDNA5 WMMA / TDM
// ---------------------------------------------------------------------------
typedef __attribute__((ext_vector_type(16))) __bf16    v16bf;
typedef __attribute__((ext_vector_type(8)))  __bf16    v8bf;
typedef __attribute__((ext_vector_type(8)))  float     v8f;
typedef __attribute__((ext_vector_type(4)))  unsigned  v4u;
typedef __attribute__((ext_vector_type(8)))  int       v8i;
typedef __attribute__((ext_vector_type(4)))  int       v4i;

#define LRELU_SLOPE 0.2f

#ifndef __has_builtin
#define __has_builtin(x) 0
#endif
#if __has_builtin(__builtin_amdgcn_tensor_load_to_lds) && __has_builtin(__builtin_amdgcn_s_wait_tensorcnt)
#define USE_TDM 1
#else
#define USE_TDM 0
#endif

// ---------------------------------------------------------------------------
// Generic utilities
// ---------------------------------------------------------------------------
__global__ void zero_kernel(float* __restrict__ p, size_t n) {
    size_t i = (size_t)blockIdx.x * blockDim.x + threadIdx.x;
    if (i < n) p[i] = 0.0f;
}

// fp32 -> bf16 (packed conversion pass)
__global__ void cvt_bf16_kernel(const float* __restrict__ in, __bf16* __restrict__ out, size_t n) {
    size_t i = (size_t)blockIdx.x * blockDim.x + threadIdx.x;
    if (i < n) out[i] = (__bf16)in[i];
}

// W[K,N] fp32 -> WT[N,K] bf16 (weights pre-transposed so GEMM B-tiles are row-major)
__global__ void cvtT_bf16_kernel(const float* __restrict__ W, __bf16* __restrict__ WT, int K, int N) {
    int i = blockIdx.x * blockDim.x + threadIdx.x;
    if (i >= K * N) return;
    int n = i / K, k = i - n * K;
    WT[(size_t)n * K + k] = (__bf16)W[(size_t)k * N + n];
}

// monotonic float <-> unsigned mapping for atomicMax on floats
__device__ __forceinline__ unsigned ford(float f) {
    unsigned u = __float_as_uint(f);
    return (u & 0x80000000u) ? ~u : (u | 0x80000000u);
}
__device__ __forceinline__ float fdec(unsigned u) {
    u = (u & 0x80000000u) ? (u & 0x7FFFFFFFu) : ~u;
    return __uint_as_float(u);
}

// ---------------------------------------------------------------------------
// TDM: load a [64 x 32] bf16 tile (row pitch K elements) into LDS with
// 16B padding per 64B row -> LDS pitch 80B (16B-aligned, bank-conflict-free).
// tensor_dim1 = total rows gives hardware OOB-zero for the ragged last tile.
// ---------------------------------------------------------------------------
#if USE_TDM
__device__ __forceinline__ void tdm_load_tile_bf16(const __bf16* gptr, unsigned ldsOff,
                                                   int K, int rowsTotal) {
    unsigned long long ga = (unsigned long long)(size_t)gptr;
    v4u g0;
    g0[0] = 1u;                                                   // count=1 (valid user D#)
    g0[1] = ldsOff;                                               // lds_addr
    g0[2] = (unsigned)(ga & 0xFFFFFFFFu);                         // global_addr lo
    g0[3] = (unsigned)((ga >> 32) & 0x01FFFFFFu) | (2u << 30);    // global_addr hi | type=2
    v8i g1;
    g1[0] = (int)((1u << 16)      // data_size = 1 (2 bytes)
                | (1u << 20)      // pad_enable
                | (3u << 22)      // pad_interval: 16 DWORDs (64B rows)
                | (3u << 25));    // pad_amount:   4 DWORDs (16B pad)
    g1[1] = (int)(((unsigned)K & 0xFFFFu) << 16);                 // tensor_dim0 lo16
    g1[2] = (int)(((unsigned)K >> 16) |
                  (((unsigned)rowsTotal & 0xFFFFu) << 16));       // dim0 hi | dim1 lo16
    g1[3] = (int)(((unsigned)rowsTotal >> 16) | (32u << 16));     // dim1 hi | tile_dim0=32
    g1[4] = 64;                                                   // tile_dim1=64, tile_dim2=0
    g1[5] = K;                                                    // tensor_dim0_stride lo32
    g1[6] = 0;
    g1[7] = 0;
    v4i gz = {0, 0, 0, 0};
#if __clang_major__ >= 23
    v8i gz8 = {0, 0, 0, 0, 0, 0, 0, 0};
    __builtin_amdgcn_tensor_load_to_lds(g0, g1, gz, gz, gz8, 0);
#else
    __builtin_amdgcn_tensor_load_to_lds(g0, g1, gz, gz, 0);
#endif
}
#endif

// ---------------------------------------------------------------------------
// WMMA bf16 GEMM: C[M,N] = act(A[M,K] @ BT[N,K]^T + bias)
// A: bf16 row-major [M,K]; BT: bf16 row-major [N,K] (pre-transposed weights).
// Block = 128 threads (4 waves). Block tile 64(M) x 64(N); K-step 32.
// Each wave: rows [w*16,w*16+16) x all 64 cols -> 4 accumulators, 4 WMMA/step.
// TDM double-buffered tile DMA when available; manual staging fallback.
// Cf (f32) and/or Cbf (bf16) outputs. act: 0=none, 1=ELU, 2=ReLU.
// ---------------------------------------------------------------------------
__global__ __launch_bounds__(128)
void gemm_wmma_bf16(const __bf16* __restrict__ A, const __bf16* __restrict__ BT,
                    const float* __restrict__ bias, float* __restrict__ Cf,
                    __bf16* __restrict__ Cbf, int M, int K, int Ncols, int act)
{
    __shared__ alignas(16) __bf16 As [2][64][40];   // 80B row pitch
    __shared__ alignas(16) __bf16 Bst[2][64][40];

    const int tid  = threadIdx.x;
    const int wave = tid >> 5;
    const int lane = tid & 31;
    const int lo   = lane & 15;
    const int hi   = lane >> 4;

    const int mBase = blockIdx.x * 64;
    const int n0    = blockIdx.y * 64;
    const int nIter = K >> 5;

    const v8f zacc = {0.f, 0.f, 0.f, 0.f, 0.f, 0.f, 0.f, 0.f};
    v8f acc[4] = {zacc, zacc, zacc, zacc};

#if USE_TDM
    const unsigned ldsA0 = (unsigned)(size_t)&As[0][0][0];
    const unsigned ldsB0 = (unsigned)(size_t)&Bst[0][0][0];
    const unsigned bufBytes = 64 * 40 * 2;
    if (wave == 0) {   // prologue: issue tile 0
        tdm_load_tile_bf16(A  + (size_t)mBase * K, ldsA0, K, M);
        tdm_load_tile_bf16(BT + (size_t)n0    * K, ldsB0, K, Ncols);
    }
#endif

    for (int it = 0; it < nIter; ++it) {
        const int k0 = it << 5;
#if USE_TDM
        if (wave == 0) {
            if (it + 1 < nIter) {   // issue next tile into other buffer, then wait tile i
                unsigned nb = (unsigned)((it + 1) & 1) * bufBytes;
                tdm_load_tile_bf16(A  + (size_t)mBase * K + k0 + 32, ldsA0 + nb, K, M);
                tdm_load_tile_bf16(BT + (size_t)n0    * K + k0 + 32, ldsB0 + nb, K, Ncols);
                __builtin_amdgcn_s_wait_tensorcnt(2);
            } else {
                __builtin_amdgcn_s_wait_tensorcnt(0);
            }
        }
        __syncthreads();
        const __bf16* asb = &As [it & 1][0][0];
        const __bf16* bsb = &Bst[it & 1][0][0];
#else
        // manual staging: 128 threads copy 64x32 bf16 tiles (16 halfs each)
        {
            int r  = tid >> 1;
            int cc = (tid & 1) * 16;
            int gr = mBase + r; if (gr >= M) gr = M - 1;
            const __bf16* ap = A + (size_t)gr * K + k0 + cc;
            if (k0 + 32 < K) __builtin_prefetch(ap + 32, 0, 1);
            *(v8bf*)&As[0][r][cc]     = *(const v8bf*)(ap);
            *(v8bf*)&As[0][r][cc + 8] = *(const v8bf*)(ap + 8);
            const __bf16* bp = BT + (size_t)(n0 + r) * K + k0 + cc;
            *(v8bf*)&Bst[0][r][cc]     = *(const v8bf*)(bp);
            *(v8bf*)&Bst[0][r][cc + 8] = *(const v8bf*)(bp + 8);
        }
        __syncthreads();
        const __bf16* asb = &As[0][0][0];
        const __bf16* bsb = &Bst[0][0][0];
#endif
        // A fragment (16-bit 16x32 layout): lane(lo,hi), a[j] = A[row][(j<8?j:j+8)+8*hi]
        const __bf16* arow = asb + (size_t)(wave * 16 + lo) * 40;
        v8bf alo = *(const v8bf*)(arow + 8 * hi);
        v8bf ahi = *(const v8bf*)(arow + 16 + 8 * hi);
        v16bf a;
        #pragma unroll
        for (int j = 0; j < 8; ++j) { a[j] = alo[j]; a[j + 8] = ahi[j]; }

        #pragma unroll
        for (int ns = 0; ns < 4; ++ns) {
            const __bf16* brow = bsb + (size_t)(ns * 16 + lo) * 40;
            v8bf blo = *(const v8bf*)(brow + 8 * hi);
            v8bf bhi = *(const v8bf*)(brow + 16 + 8 * hi);
            v16bf b;
            #pragma unroll
            for (int j = 0; j < 8; ++j) { b[j] = blo[j]; b[j + 8] = bhi[j]; }
            acc[ns] = __builtin_amdgcn_wmma_f32_16x16x32_bf16(
                false, a, false, b, (short)0, acc[ns], false, false);
        }
        __syncthreads();   // all reads done before next DMA/staging overwrites
    }

    // store: VGPR r -> row = tile + 8*hi + r; lane lo -> col within 16-wide group
    const int rowTile = mBase + wave * 16;
    if (rowTile < M) {
        const int row0 = rowTile + 8 * hi;
        #pragma unroll
        for (int ns = 0; ns < 4; ++ns) {
            const int col = n0 + ns * 16 + lo;
            const float bv = bias ? bias[col] : 0.0f;
            #pragma unroll
            for (int r = 0; r < 8; ++r) {
                float v = acc[ns][r] + bv;
                if (act == 1)      v = (v > 0.f) ? v : (__expf(v) - 1.f);
                else if (act == 2) v = (v > 0.f) ? v : 0.f;
                if (Cf)  Cf [(size_t)(row0 + r) * Ncols + col] = v;
                if (Cbf) Cbf[(size_t)(row0 + r) * Ncols + col] = (__bf16)v;
            }
        }
    }
}

// ---------------------------------------------------------------------------
// Self-loop edge_attr: per-dst mean of incoming edge_attr (fill_value='mean')
// ---------------------------------------------------------------------------
__global__ void loop_accum_kernel(const int* __restrict__ dst, const float* __restrict__ eattr,
                                  float* __restrict__ lsum, float* __restrict__ lcnt, int E) {
    int e = blockIdx.x * blockDim.x + threadIdx.x;
    if (e >= E) return;
    int d = dst[e];
    const float* ea = eattr + (size_t)e * 16;
    #pragma unroll
    for (int f = 0; f < 16; ++f) atomicAdd(&lsum[(size_t)d * 16 + f], ea[f]);
    atomicAdd(&lcnt[d], 1.0f);
}

__global__ void loop_div_kernel(float* __restrict__ lsum, const float* __restrict__ lcnt, int NV) {
    int i = blockIdx.x * blockDim.x + threadIdx.x;
    if (i >= NV * 16) return;
    lsum[i] = lsum[i] / fmaxf(lcnt[i >> 4], 1.0f);
}

// v[f*8+h] = sum_c We[f*256 + h*32 + c] * ae[h*32 + c]
__global__ void vcompute_kernel(const float* __restrict__ We, const float* __restrict__ ae,
                                float* __restrict__ v) {
    int i = threadIdx.x;
    if (i < 128) {
        int f = i >> 3, h = i & 7;
        float s = 0.f;
        #pragma unroll
        for (int c = 0; c < 32; ++c) s += We[(size_t)f * 256 + h * 32 + c] * ae[h * 32 + c];
        v[f * 8 + h] = s;
    }
}

// Per-node attention terms: one wave32 per node, shuffle reduction
__global__ void node_att_kernel(const float* __restrict__ xs,
                                const float* __restrict__ att_s, const float* __restrict__ att_d,
                                float* __restrict__ as_out, float* __restrict__ ad_out, int NV) {
    int node = blockIdx.x * (blockDim.x >> 5) + (threadIdx.x >> 5);
    if (node >= NV) return;
    int lane = threadIdx.x & 31;
    #pragma unroll
    for (int h = 0; h < 8; ++h) {
        float p  = xs[(size_t)node * 256 + h * 32 + lane];
        float vs = p * att_s[h * 32 + lane];
        float vd = p * att_d[h * 32 + lane];
        #pragma unroll
        for (int o = 16; o > 0; o >>= 1) {
            vs += __shfl_xor(vs, o, 32);
            vd += __shfl_xor(vd, o, 32);
        }
        if (lane == 0) { as_out[node * 8 + h] = vs; ad_out[node * 8 + h] = vd; }
    }
}

__device__ __forceinline__ float edge_logit(const float* __restrict__ ea, const float* __restrict__ v,
                                            const float* __restrict__ as_, const float* __restrict__ ad_,
                                            int s, int d, int h) {
    float a = as_[(size_t)s * 8 + h] + ad_[(size_t)d * 8 + h];
    #pragma unroll
    for (int f = 0; f < 16; ++f) a += ea[f] * v[f * 8 + h];
    return (a > 0.f) ? a : LRELU_SLOPE * a;
}

__global__ void pass_max_kernel(const int* __restrict__ src, const int* __restrict__ dst,
                                const float* __restrict__ eattr, const float* __restrict__ lattr,
                                const float* __restrict__ v, const float* __restrict__ as_,
                                const float* __restrict__ ad_, unsigned* __restrict__ m,
                                int E, int NV) {
    int e = blockIdx.x * blockDim.x + threadIdx.x;
    if (e >= E + NV) return;
    int s, d; const float* ea;
    if (e < E) { s = src[e]; d = dst[e]; ea = eattr + (size_t)e * 16; }
    else       { s = e - E;  d = s;      ea = lattr + (size_t)s * 16; }
    #pragma unroll
    for (int h = 0; h < 8; ++h) {
        float a = edge_logit(ea, v, as_, ad_, s, d, h);
        atomicMax(&m[(size_t)d * 8 + h], ford(a));
    }
}

__global__ void pass_sum_kernel(const int* __restrict__ src, const int* __restrict__ dst,
                                const float* __restrict__ eattr, const float* __restrict__ lattr,
                                const float* __restrict__ v, const float* __restrict__ as_,
                                const float* __restrict__ ad_, const unsigned* __restrict__ m,
                                float* __restrict__ den, int E, int NV) {
    int e = blockIdx.x * blockDim.x + threadIdx.x;
    if (e >= E + NV) return;
    int s, d; const float* ea;
    if (e < E) { s = src[e]; d = dst[e]; ea = eattr + (size_t)e * 16; }
    else       { s = e - E;  d = s;      ea = lattr + (size_t)s * 16; }
    #pragma unroll
    for (int h = 0; h < 8; ++h) {
        float a  = edge_logit(ea, v, as_, ad_, s, d, h);
        atomicAdd(&den[(size_t)d * 8 + h], __expf(a - fdec(m[(size_t)d * 8 + h])));
    }
}

// One wave32 per edge; lane handles 8 consecutive channels (one head each)
__global__ void pass_agg_kernel(const int* __restrict__ src, const int* __restrict__ dst,
                                const float* __restrict__ eattr, const float* __restrict__ lattr,
                                const float* __restrict__ v, const float* __restrict__ as_,
                                const float* __restrict__ ad_, const unsigned* __restrict__ m,
                                const float* __restrict__ den, const float* __restrict__ xs,
                                float* __restrict__ agg, int E, int NV) {
    int wid = blockIdx.x * (blockDim.x >> 5) + (threadIdx.x >> 5);
    if (wid >= E + NV) return;
    int lane = threadIdx.x & 31;
    int s, d; const float* ea;
    if (wid < E) { s = src[wid]; d = dst[wid]; ea = eattr + (size_t)wid * 16; }
    else         { s = wid - E;  d = s;        ea = lattr + (size_t)s * 16; }
    int h = lane >> 2;
    float a = edge_logit(ea, v, as_, ad_, s, d, h);
    float w = __expf(a - fdec(m[(size_t)d * 8 + h])) / (den[(size_t)d * 8 + h] + 1e-16f);
    const float* xr = xs  + (size_t)s * 256 + lane * 8;
    float*       ow = agg + (size_t)d * 256 + lane * 8;
    __builtin_prefetch(xr, 0, 1);
    #pragma unroll
    for (int j = 0; j < 8; ++j) atomicAdd(&ow[j], xr[j] * w);
}

// layer-1 finalize: h1 = elu(agg + b1), written directly as bf16 GEMM input
__global__ void bias_elu_bf16_kernel(const float* __restrict__ aggv, const float* __restrict__ bias,
                                     __bf16* __restrict__ outbf, size_t n, int Cdim) {
    size_t i = (size_t)blockIdx.x * blockDim.x + threadIdx.x;
    if (i >= n) return;
    float v = aggv[i] + bias[i % Cdim];
    v = (v > 0.f) ? v : (__expf(v) - 1.f);
    outbf[i] = (__bf16)v;
}

// layer-2 finalize: mean over heads + b2 + elu
__global__ void head_mean_kernel(const float* __restrict__ aggv, const float* __restrict__ b2,
                                 float* __restrict__ h2, int NV) {
    int i = blockIdx.x * blockDim.x + threadIdx.x;
    if (i >= NV * 32) return;
    int n = i >> 5, c = i & 31;
    float s = 0.f;
    #pragma unroll
    for (int h = 0; h < 8; ++h) s += aggv[(size_t)n * 256 + h * 32 + c];
    s = s * 0.125f + b2[c];
    h2[i] = (s > 0.f) ? s : (__expf(s) - 1.f);
}

// global mean pool + concat u -> bf16 MLP input [NB, 96]
__global__ void pool_kernel(const float* __restrict__ h2, const float* __restrict__ u,
                            __bf16* __restrict__ comb, int npg) {
    int g = blockIdx.x, t = threadIdx.x;   // 96 threads
    if (t < 32) {
        float s = 0.f;
        for (int i = 0; i < npg; ++i) s += h2[(size_t)(g * npg + i) * 32 + t];
        comb[(size_t)g * 96 + t] = (__bf16)(s / (float)npg);
    } else if (t < 96) {
        comb[(size_t)g * 96 + t] = (__bf16)u[(size_t)g * 64 + (t - 32)];
    }
}

__global__ void gemv_final_kernel(const float* __restrict__ A, const float* __restrict__ w,
                                  const float* __restrict__ b, float* __restrict__ out,
                                  int M, int K) {
    int r = blockIdx.x * blockDim.x + threadIdx.x;
    if (r >= M) return;
    float s = b[0];
    for (int k = 0; k < K; ++k) s += A[(size_t)r * K + k] * w[k];
    out[r] = s;
}

// ---------------------------------------------------------------------------
// Launch
// ---------------------------------------------------------------------------
extern "C" void kernel_launch(void* const* d_in, const int* in_sizes, int n_in,
                              void* d_out, int out_size, void* d_ws, size_t ws_size,
                              hipStream_t stream) {
    const float* x     = (const float*)d_in[0];
    const int*   eidx  = (const int*)  d_in[1];
    const float* eattr = (const float*)d_in[2];
    const float* u     = (const float*)d_in[3];
    /* d_in[4] = batch: unused (graphs are equal-size contiguous) */
    const float* W1  = (const float*)d_in[5];
    const float* as1 = (const float*)d_in[6];
    const float* ad1 = (const float*)d_in[7];
    const float* We1 = (const float*)d_in[8];
    const float* ae1 = (const float*)d_in[9];
    const float* b1  = (const float*)d_in[10];
    const float* W2  = (const float*)d_in[11];
    const float* as2 = (const float*)d_in[12];
    const float* ad2 = (const float*)d_in[13];
    const float* We2 = (const float*)d_in[14];
    const float* ae2 = (const float*)d_in[15];
    const float* b2  = (const float*)d_in[16];
    const float* mW1 = (const float*)d_in[17];
    const float* mb1 = (const float*)d_in[18];
    const float* mW2 = (const float*)d_in[19];
    const float* mb2 = (const float*)d_in[20];
    const float* mW3 = (const float*)d_in[21];
    const float* mb3 = (const float*)d_in[22];
    float* out = (float*)d_out;
    (void)n_in; (void)out_size; (void)ws_size;

    const int NV  = in_sizes[0] / 128;   // 50000 nodes
    const int E   = in_sizes[2] / 16;    // 400000 edges
    const int NB  = in_sizes[3] / 64;    // 2000 graphs
    const int npg = NV / NB;             // 25 nodes/graph
    const int ET  = E + NV;

    const int* src = eidx;
    const int* dst = eidx + E;

    // ---- workspace carve-up ----
    char* ws = (char*)d_ws;
    size_t off = 0;
    auto alloc = [&](size_t bytes) -> void* {
        void* p = ws + off;
        off = (off + bytes + 255) & ~(size_t)255;
        return p;
    };
    float*    xs   = (float*)   alloc((size_t)NV * 256 * 4);
    float*    agg  = (float*)   alloc((size_t)NV * 256 * 4);
    __bf16*   abf  = (__bf16*)  alloc((size_t)NV * 256 * 2);   // xbf / h1bf / m1bf (sequential reuse)
    float*    as_b = (float*)   alloc((size_t)NV * 8 * 4);
    float*    ad_b = (float*)   alloc((size_t)NV * 8 * 4);
    unsigned* mbuf = (unsigned*)alloc((size_t)NV * 8 * 4);
    float*    den  = (float*)   alloc((size_t)NV * 8 * 4);
    float*    lsum = (float*)   alloc((size_t)NV * 16 * 4);    // becomes loop_attr in place
    float*    lcnt = (float*)   alloc((size_t)NV * 4);
    float*    vbuf = (float*)   alloc(128 * 4);
    float*    h2   = (float*)   alloc((size_t)NV * 32 * 4);
    __bf16*   comb = (__bf16*)  alloc((size_t)NB * 96 * 2);
    __bf16*   wbft = (__bf16*)  alloc((size_t)256 * 256 * 2);  // transposed-weight buffer (reused)
    float*    m2b  = (float*)   alloc((size_t)NB * 128 * 4);

    auto zero = [&](void* p, size_t nf) {
        zero_kernel<<<(unsigned)((nf + 255) / 256), 256, 0, stream>>>((float*)p, nf);
    };
    auto cvtT = [&](const float* W, int K, int N) {
        cvtT_bf16_kernel<<<(K * N + 255) / 256, 256, 0, stream>>>(W, wbft, K, N);
    };
    auto gemm = [&](const __bf16* A, float* Cf, __bf16* Cbf, const float* bias,
                    int M, int K, int Ncols, int act) {
        dim3 g((M + 63) / 64, Ncols / 64);
        gemm_wmma_bf16<<<g, 128, 0, stream>>>(A, wbft, bias, Cf, Cbf, M, K, Ncols, act);
    };

    const unsigned ebl = (ET + 255) / 256;   // thread-per-edge grids
    const unsigned wbl = (ET + 7) / 8;       // wave-per-edge grids (8 waves/block)

    // ---- self-loop edge_attr = per-dst mean ----
    zero(lsum, (size_t)NV * 16);
    zero(lcnt, (size_t)NV);
    loop_accum_kernel<<<(E + 255) / 256, 256, 0, stream>>>(dst, eattr, lsum, lcnt, E);
    loop_div_kernel<<<(NV * 16 + 255) / 256, 256, 0, stream>>>(lsum, lcnt, NV);

    // =================== GAT layer 1 (concat heads) ===================
    cvt_bf16_kernel<<<(unsigned)(((size_t)NV * 128 + 255) / 256), 256, 0, stream>>>(x, abf, (size_t)NV * 128);
    cvtT(W1, 128, 256);
    gemm(abf, xs, nullptr, nullptr, NV, 128, 256, 0);             // xs = x @ W1 (TDM + WMMA)
    node_att_kernel<<<(NV + 7) / 8, 256, 0, stream>>>(xs, as1, ad1, as_b, ad_b, NV);
    vcompute_kernel<<<1, 128, 0, stream>>>(We1, ae1, vbuf);
    zero(mbuf, (size_t)NV * 8);
    zero(den,  (size_t)NV * 8);
    zero(agg,  (size_t)NV * 256);
    pass_max_kernel<<<ebl, 256, 0, stream>>>(src, dst, eattr, lsum, vbuf, as_b, ad_b, mbuf, E, NV);
    pass_sum_kernel<<<ebl, 256, 0, stream>>>(src, dst, eattr, lsum, vbuf, as_b, ad_b, mbuf, den, E, NV);
    pass_agg_kernel<<<wbl, 256, 0, stream>>>(src, dst, eattr, lsum, vbuf, as_b, ad_b, mbuf, den, xs, agg, E, NV);
    bias_elu_bf16_kernel<<<(unsigned)(((size_t)NV * 256 + 255) / 256), 256, 0, stream>>>(
        agg, b1, abf, (size_t)NV * 256, 256);                     // abf = h1 (bf16)

    // =================== GAT layer 2 (mean heads) ===================
    cvtT(W2, 256, 256);
    gemm(abf, xs, nullptr, nullptr, NV, 256, 256, 0);             // xs2 = h1 @ W2
    node_att_kernel<<<(NV + 7) / 8, 256, 0, stream>>>(xs, as2, ad2, as_b, ad_b, NV);
    vcompute_kernel<<<1, 128, 0, stream>>>(We2, ae2, vbuf);
    zero(mbuf, (size_t)NV * 8);
    zero(den,  (size_t)NV * 8);
    zero(agg,  (size_t)NV * 256);
    pass_max_kernel<<<ebl, 256, 0, stream>>>(src, dst, eattr, lsum, vbuf, as_b, ad_b, mbuf, E, NV);
    pass_sum_kernel<<<ebl, 256, 0, stream>>>(src, dst, eattr, lsum, vbuf, as_b, ad_b, mbuf, den, E, NV);
    pass_agg_kernel<<<wbl, 256, 0, stream>>>(src, dst, eattr, lsum, vbuf, as_b, ad_b, mbuf, den, xs, agg, E, NV);
    head_mean_kernel<<<(NV * 32 + 255) / 256, 256, 0, stream>>>(agg, b2, h2, NV);

    // =================== pool + MLP ===================
    pool_kernel<<<NB, 96, 0, stream>>>(h2, u, comb, npg);
    cvtT(mW1, 96, 256);
    gemm(comb, nullptr, abf, mb1, NB, 96, 256, 2);                // m1 (ReLU) -> bf16
    cvtT(mW2, 256, 128);
    gemm(abf, m2b, nullptr, mb2, NB, 256, 128, 2);                // m2 (ReLU) -> f32
    gemv_final_kernel<<<(NB + 255) / 256, 256, 0, stream>>>(m2b, mW3, mb3, out, NB, 128);
}
